// SOSFlow_48438641164546
// MI455X (gfx1250) — compile-verified, matrix operand order checked by reading, not run on hardware
//
#include <hip/hip_runtime.h>

// ---------------------------------------------------------------------------
// SOSFlow fused pipeline for gfx1250 (MI455X, wave32, WMMA)
//   GEMM1 (x@W1+b1, relu)  -> bf16 h in ws            [v_wmma_f32_16x16x32_bf16]
//   GEMM2 (h@W2+b2) + polynomial epilogue, A staged in LDS via
//   global_load_async_to_lds_b128 (ASYNCcnt), B streamed from L2-resident W2t.
//   Deterministic logdet reduction (no float atomics).
// ---------------------------------------------------------------------------

typedef __bf16 bf16_t;
typedef __attribute__((ext_vector_type(16))) __bf16    v16bf;
typedef __attribute__((ext_vector_type(2)))  __bf16    v2bf;
typedef __attribute__((ext_vector_type(8)))  float     v8f;
typedef __attribute__((ext_vector_type(4)))  float     f32x4;
typedef __attribute__((ext_vector_type(4)))  unsigned  u32x4;

#define B_ROWS   8192
#define D_DIM    64
#define H_DIM    512
#define OUT_COLS 1664   // D_DIM * (K*M + 1) = 64 * 26

#if defined(__has_builtin)
#if __has_builtin(__builtin_amdgcn_cvt_pk_bf16_f32)
#define HAVE_CVT_PK_BF16 1
#endif
#endif

__device__ __forceinline__ bf16_t f2bf(float f) {
  // round-to-nearest-even f32 -> bf16 (scalar fallback)
  unsigned u = __builtin_bit_cast(unsigned, f);
  unsigned r = u + 0x7FFFu + ((u >> 16) & 1u);
  unsigned short s = (unsigned short)(r >> 16);
  return __builtin_bit_cast(bf16_t, s);
}

__device__ __forceinline__ v2bf f2bf2(float a, float b) {
#ifdef HAVE_CVT_PK_BF16
  auto t = __builtin_amdgcn_cvt_pk_bf16_f32(a, b);   // v_cvt_pk_bf16_f32
  return __builtin_bit_cast(v2bf, t);
#else
  v2bf r; r[0] = f2bf(a); r[1] = f2bf(b); return r;
#endif
}

// ---------------------------------------------------------------------------
// Prep: W1 (64x512) -> W1t bf16 (n-major, 512x64)
//       W2 (512x1664) -> W2t bf16 (n-major, 1664x512)
// n-major layout makes each WMMA B-fragment two contiguous 16B loads per lane.
// ---------------------------------------------------------------------------
__global__ void prep_kernel(const float* __restrict__ W1, const float* __restrict__ W2,
                            bf16_t* __restrict__ W1t, bf16_t* __restrict__ W2t) {
  int idx    = blockIdx.x * blockDim.x + threadIdx.x;
  int stride = gridDim.x * blockDim.x;
  for (int i = idx; i < H_DIM * D_DIM; i += stride) {
    int n = i / D_DIM, k = i % D_DIM;
    W1t[i] = f2bf(W1[k * H_DIM + n]);
  }
  for (int i = idx; i < OUT_COLS * H_DIM; i += stride) {
    int n = i / H_DIM, k = i % H_DIM;
    W2t[i] = f2bf(W2[k * OUT_COLS + n]);
  }
}

// ---------------------------------------------------------------------------
// GEMM1: h = relu(x @ W1 + b1), h stored bf16 row-major (8192 x 512).
// One 16x16 C-tile per wave; K=64 -> two 16x16x32 bf16 WMMAs.
// ---------------------------------------------------------------------------
__global__ __launch_bounds__(256)
void gemm1_kernel(const float* __restrict__ x, const bf16_t* __restrict__ W1t,
                  const float* __restrict__ b1, bf16_t* __restrict__ hout) {
  const int wid     = blockIdx.x * (blockDim.x >> 5) + (threadIdx.x >> 5);
  const int lane    = threadIdx.x & 31;
  const int rowTile = wid >> 5;        // 512 row tiles
  const int nTile   = wid & 31;        // 32 n tiles
  const int row     = rowTile * 16 + (lane & 15);
  const int n       = nTile * 16 + (lane & 15);
  const int khalf   = (lane >> 4) * 8; // K packing: lanes 0-15 -> K 0..7/16..23

  v8f acc = {};
#pragma unroll
  for (int ks = 0; ks < 2; ++ks) {
    const int kb = ks * 32 + khalf;
    // A fragment: x row f32 -> bf16 via packed converts; two runs of 8
    const float* ap0 = x + row * D_DIM + kb;   // 32B-aligned
    f32x4 f0 = *(const f32x4*)(ap0);
    f32x4 f1 = *(const f32x4*)(ap0 + 4);
    f32x4 f2 = *(const f32x4*)(ap0 + 16);
    f32x4 f3 = *(const f32x4*)(ap0 + 20);
    union { v16bf v; v2bf p[8]; } a;
    a.p[0] = f2bf2(f0[0], f0[1]); a.p[1] = f2bf2(f0[2], f0[3]);
    a.p[2] = f2bf2(f1[0], f1[1]); a.p[3] = f2bf2(f1[2], f1[3]);
    a.p[4] = f2bf2(f2[0], f2[1]); a.p[5] = f2bf2(f2[2], f2[3]);
    a.p[6] = f2bf2(f3[0], f3[1]); a.p[7] = f2bf2(f3[2], f3[3]);
    // B fragment: W1t column n, contiguous bf16 runs (16B vector loads)
    union { v16bf v; u32x4 q[2]; } b;
    const bf16_t* bp = W1t + n * D_DIM + kb;
    b.q[0] = *(const u32x4*)(bp);
    b.q[1] = *(const u32x4*)(bp + 16);
    acc = __builtin_amdgcn_wmma_f32_16x16x32_bf16(false, a.v, false, b.v,
                                                  (short)0, acc, false, false);
  }
  const float bias = b1[n];
#pragma unroll
  for (int r = 0; r < 8; ++r) {
    const int m = rowTile * 16 + r + 8 * (lane >> 4); // C/D: M = r + 8*(lane>=16)
    float v = acc[r] + bias;
    v = v > 0.f ? v : 0.f;
    hout[m * H_DIM + n] = f2bf(v);
  }
}

// ---------------------------------------------------------------------------
// GEMM2 + fused epilogue.
// Block: 128 threads = 4 waves, covers 16 rows x 832 cols (32 output dims).
// Wave w owns 13 N-tiles = 208 cols = exactly 8 dims x 26 params.
// A strip (16x512 bf16 = 16KB) staged ONCE per block into LDS via
// global_load_async_to_lds_b128; fragments then come from ds_load_b128.
// The staging region aliases the params-spill LDS (phases separated by
// barriers), keeping the block at 53KB LDS.
// ---------------------------------------------------------------------------
__global__ __launch_bounds__(128)
void gemm2_epilogue_kernel(const bf16_t* __restrict__ h, const bf16_t* __restrict__ W2t,
                           const float* __restrict__ b2, const float* __restrict__ x,
                           float* __restrict__ Zout, float* __restrict__ logPartial) {
  __shared__ __align__(16) float sp[4 * 16 * 208];  // 53248 B
  bf16_t* shh = (bf16_t*)sp;                        // first 16KB: staged h strip

  const int w       = threadIdx.x >> 5;
  const int lane    = threadIdx.x & 31;
  const int tid     = threadIdx.x;
  const int rt      = blockIdx.x >> 1;  // 512 row tiles
  const int c       = blockIdx.x & 1;   // 2 column chunks of 32 dims
  const int rowBase = rt * 16;
  const int dBase   = c * 32 + w * 8;
  const int colBase = dBase * 26;       // 26-aligned: wave owns whole dims
  const int khalf   = (lane >> 4) * 8;

  // ---- async stage: 16KB contiguous (rows rowBase..rowBase+15 of h) ----
  {
    const char* gsrc = (const char*)(h + (size_t)rowBase * H_DIM) + tid * 128;
    unsigned long long gaddr = (unsigned long long)(uintptr_t)gsrc;
    unsigned ldsa = (unsigned)(uintptr_t)(shh) + (unsigned)(tid * 128);
    asm volatile("global_load_async_to_lds_b128 %0, %1, off offset:0"   :: "v"(ldsa), "v"(gaddr) : "memory");
    asm volatile("global_load_async_to_lds_b128 %0, %1, off offset:16"  :: "v"(ldsa), "v"(gaddr) : "memory");
    asm volatile("global_load_async_to_lds_b128 %0, %1, off offset:32"  :: "v"(ldsa), "v"(gaddr) : "memory");
    asm volatile("global_load_async_to_lds_b128 %0, %1, off offset:48"  :: "v"(ldsa), "v"(gaddr) : "memory");
    asm volatile("global_load_async_to_lds_b128 %0, %1, off offset:64"  :: "v"(ldsa), "v"(gaddr) : "memory");
    asm volatile("global_load_async_to_lds_b128 %0, %1, off offset:80"  :: "v"(ldsa), "v"(gaddr) : "memory");
    asm volatile("global_load_async_to_lds_b128 %0, %1, off offset:96"  :: "v"(ldsa), "v"(gaddr) : "memory");
    asm volatile("global_load_async_to_lds_b128 %0, %1, off offset:112" :: "v"(ldsa), "v"(gaddr) : "memory");
    asm volatile("s_wait_asynccnt 0x0" ::: "memory");
  }
  __syncthreads();

  v8f acc[13] = {};
  const int lrow = lane & 15;

  for (int ks = 0; ks < 16; ++ks) {
    const int kb = ks * 32 + khalf;
    // A fragment from LDS (ds_load_b128 x2)
    union { v16bf v; u32x4 q[2]; } a;
    const bf16_t* ap = shh + lrow * H_DIM + kb;
    a.q[0] = *(const u32x4*)(ap);
    a.q[1] = *(const u32x4*)(ap + 16);
    if (ks < 15)  // keep next k-step's B stream warm (global_prefetch_b8)
      __builtin_prefetch((const void*)(W2t + (colBase + (lane & 15)) * H_DIM + kb + 32), 0, 0);
#pragma unroll
    for (int t = 0; t < 13; ++t) {
      const int n = colBase + t * 16 + (lane & 15);
      union { v16bf v; u32x4 q[2]; } b;
      const bf16_t* bp = W2t + n * H_DIM + kb;
      b.q[0] = *(const u32x4*)(bp);
      b.q[1] = *(const u32x4*)(bp + 16);
      acc[t] = __builtin_amdgcn_wmma_f32_16x16x32_bf16(false, a.v, false, b.v,
                                                       (short)0, acc[t], false, false);
    }
  }

  // all waves done reading the staged h strip before we overwrite it
  __syncthreads();

  // bias + spill params tile to this wave's LDS slab
  float* slab = sp + w * (16 * 208);
#pragma unroll
  for (int t = 0; t < 13; ++t) {
    const int nl   = t * 16 + (lane & 15);
    const float bv = b2[colBase + nl];
#pragma unroll
    for (int r = 0; r < 8; ++r) {
      const int m = r + 8 * (lane >> 4);
      slab[m * 208 + nl] = acc[t][r] + bv;
    }
  }
  __syncthreads();

  // Epilogue: 16 rows x 8 dims per wave = 128 items, 4 per lane.
  // invf[m] = 1/(i+j+1) for m = i+j
  const float invf[9] = {1.f, 0.5f, 1.f/3.f, 0.25f, 0.2f,
                         1.f/6.f, 1.f/7.f, 0.125f, 1.f/9.f};
  const int rowG = rowBase + lrow;
  float plog = 0.f;
#pragma unroll
  for (int it = 0; it < 4; ++it) {
    const int dl = (lane >> 4) + 2 * it;          // 0..7
    const int dG = dBase + dl;
    const float z = x[rowG * D_DIM + dG];
    float X[5];
    X[0] = 1.f; X[1] = z; X[2] = z * z; X[3] = X[2] * z; X[4] = X[2] * X[2];
    const float* p = sp + w * (16 * 208) + lrow * 208 + dl * 26;
    float sfilt = 0.f, t2 = 0.f;
#pragma unroll
    for (int k = 0; k < 5; ++k) {
      float u[5]; float dot = 0.f;
#pragma unroll
      for (int i = 0; i < 5; ++i) { u[i] = p[k * 5 + i] * X[i]; dot += u[i]; }
      t2 += dot * dot;                             // transform(C) = sum_k (c_k.X)^2
#pragma unroll
      for (int i = 0; i < 5; ++i) {
        float a2 = 0.f;
#pragma unroll
        for (int j = 0; j < 5; ++j) a2 += u[j] * invf[i + j];
        sfilt += u[i] * a2;                        // transform(C/filt)
      }
    }
    Zout[rowG * D_DIM + dG] = sfilt * z + p[25];
    plog += __logf(fabsf(t2));
  }
  // deterministic partial: unique slot per (row, chunk, wave, lane-half)
  logPartial[rowG * 16 + c * 8 + w * 2 + (lane >> 4)] = plog;
}

// ---------------------------------------------------------------------------
// Finalize: logdet[row] = sum of 16 partials.
// ---------------------------------------------------------------------------
__global__ void finalize_kernel(const float* __restrict__ logPartial,
                                float* __restrict__ logdet) {
  const int r = blockIdx.x * blockDim.x + threadIdx.x;
  if (r < B_ROWS) {
    float s = 0.f;
#pragma unroll
    for (int i = 0; i < 16; ++i) s += logPartial[r * 16 + i];
    logdet[r] = s;
  }
}

// ---------------------------------------------------------------------------
extern "C" void kernel_launch(void* const* d_in, const int* in_sizes, int n_in,
                              void* d_out, int out_size, void* d_ws, size_t ws_size,
                              hipStream_t stream) {
  const float* x  = (const float*)d_in[0];   // (8192, 64)
  const float* W1 = (const float*)d_in[1];   // (64, 512)
  const float* b1 = (const float*)d_in[2];   // (512,)
  const float* W2 = (const float*)d_in[3];   // (512, 1664)
  const float* b2 = (const float*)d_in[4];   // (1664,)

  float* out    = (float*)d_out;
  float* Zout   = out;                        // (8192, 64)
  float* logdet = out + B_ROWS * D_DIM;       // (8192, 1)

  char* ws = (char*)d_ws;
  bf16_t* W1t        = (bf16_t*)(ws);                                   //   64 KB
  bf16_t* W2t        = (bf16_t*)(ws + 65536);                           // 1664 KB
  bf16_t* hbuf       = (bf16_t*)(ws + 65536 + 1703936);                 // 8192 KB
  float*  logPartial = (float* )(ws + 65536 + 1703936 + 8388608);       //  512 KB

  prep_kernel<<<1024, 256, 0, stream>>>(W1, W2, W1t, W2t);
  gemm1_kernel<<<2048, 256, 0, stream>>>(x, W1t, b1, hbuf);
  gemm2_epilogue_kernel<<<1024, 128, 0, stream>>>(hbuf, W2t, b2, x, Zout, logPartial);
  finalize_kernel<<<(B_ROWS + 255) / 256, 256, 0, stream>>>(logPartial, logdet);
}
